// TGNLayerGraphAttentionEmbedding_48747878810095
// MI455X (gfx1250) — matrix operand order repaired and checked by reading, not compiled, not run on hardware
//
#include <hip/hip_runtime.h>
#include <cstdint>
#include <cstddef>

// ---------------------------------------------------------------------------
// TGN graph-attention layer for MI455X (gfx1250, wave32, WMMA).
// - All matmuls on v_wmma_f32_16x16x32_bf16 (bf16 in, f32 accumulate).
// - Flash attention (online softmax, no NxN score materialization).
// - K/V/Q projection GEMM: 16x192 row-strip per wave; B slice staged in LDS
//   via the Tensor Data Mover (tensor_load_to_lds), double-buffered,
//   synchronized with s_wait_tensorcnt.
// ---------------------------------------------------------------------------

typedef __attribute__((ext_vector_type(16))) __bf16 v16bf;
typedef __attribute__((ext_vector_type(8)))  float  v8f;
typedef __attribute__((ext_vector_type(4)))  unsigned int u32x4;
typedef __attribute__((ext_vector_type(8)))  int i32x8;
typedef __attribute__((ext_vector_type(4)))  int i32x4;

using u16 = unsigned short;
using u32 = unsigned int;
using u64 = unsigned long long;

struct alignas(16) V4 { u32 x, y, z, w; };
union Frag { v16bf bf; V4 q[2]; };   // 32 bytes = one wave32 WMMA bf16 operand slice per lane

// round-to-nearest-even f32 -> bf16
__device__ __forceinline__ u16 f2bf(float f) {
  union { float f; u32 u; } c; c.f = f;
  u32 r = c.u + 0x7FFFu + ((c.u >> 16) & 1u);
  return (u16)(r >> 16);
}

// Problem constants (match reference)
#define NN    4096
#define KNB   10
#define EE    128
#define EFD   64
#define TD    64
#define HH    2
#define DD    192          // E + T
#define KEYD  256          // E + EF + T
#define KDIM  2560         // KEYD * KNB
#define DHEAD 96           // D / H
#define DHE   320          // D + E

#if defined(__has_builtin)
#  if __has_builtin(__builtin_amdgcn_tensor_load_to_lds)
#    define HAVE_TDM 1
#  else
#    define HAVE_TDM 0
#  endif
#else
#  define HAVE_TDM 0
#endif

// ---------------------------------------------------------------------------
// Prep kernels
// ---------------------------------------------------------------------------

// agg[n][kk*256 + j] = concat(features[neighbors[n][kk]], edge[n][kk], time[n][kk]) -> bf16
__global__ void k_build_agg(const float* __restrict__ features,
                            const float* __restrict__ edgef,
                            const float* __restrict__ timef,
                            const int*   __restrict__ neigh,
                            u16* __restrict__ agg) {
  long idx = (long)blockIdx.x * blockDim.x + threadIdx.x;
  long total = (long)NN * KDIM;
  if (idx >= total) return;
  int n  = (int)(idx / KDIM);
  int r  = (int)(idx % KDIM);
  int kk = r / KEYD;
  int j  = r % KEYD;
  float v;
  if (j < EE) {
    int src = neigh[n * KNB + kk];
    v = features[(long)src * EE + j];
  } else if (j < EE + EFD) {
    v = edgef[((long)n * KNB + kk) * EFD + (j - EE)];
  } else {
    v = timef[((long)n * KNB + kk) * TD + (j - EE - EFD)];
  }
  agg[idx] = f2bf(v);
}

// q_in[n][0:128]=features[node_idx[n]], q_in[n][128:192]=0 ; also h[:,0:128]=features (bf16)
__global__ void k_build_qin_h(const float* __restrict__ features,
                              const int*   __restrict__ nidx,
                              u16* __restrict__ qin,
                              u16* __restrict__ hbuf) {
  long idx = (long)blockIdx.x * blockDim.x + threadIdx.x;
  long total = (long)NN * DD;
  if (idx >= total) return;
  int n = (int)(idx / DD);
  int j = (int)(idx % DD);
  float v = 0.f;
  if (j < EE) {
    v = features[(long)nidx[n] * EE + j];
    hbuf[(long)n * DHE + j] = f2bf(v);
  }
  qin[idx] = f2bf(v);
}

// Wt[j][k] = W[k][j]  (W is [K x Ncols] f32 row-major; Wt is [Ncols x K] bf16 row-major)
__global__ void k_transpose_w(const float* __restrict__ W, u16* __restrict__ Wt,
                              int K, int Ncols) {
  long idx = (long)blockIdx.x * blockDim.x + threadIdx.x;
  long total = (long)K * Ncols;
  if (idx >= total) return;
  int k = (int)(idx / Ncols);
  int j = (int)(idx % Ncols);
  Wt[(long)j * K + k] = f2bf(W[idx]);
}

// ---------------------------------------------------------------------------
// TDM helper: stage a [DD rows x 32 cols] bf16 slice of Bt (row stride Kd
// elements) into LDS at byte offset lds_off.  2-D tile descriptor per ISA
// CDNA5 ISA (D# group 0 + group 1; groups 2/3 zero for 2-D).
// This toolchain's builtin takes 6 args (g0, g1, g2, g3, extra, cpol).
// ---------------------------------------------------------------------------
#if HAVE_TDM
__device__ __forceinline__ void tdm_stage_b(const u16* Bt, int Kd, int k0, u32 lds_off) {
  u64 ga = (u64)(uintptr_t)(Bt + (size_t)k0);       // tile start (row 0, col k0)
  u32x4 g0;
  g0[0] = 1u;                                        // count=1 (valid user descriptor)
  g0[1] = lds_off;                                   // lds_addr [63:32]
  g0[2] = (u32)ga;                                   // global_addr [95:64]
  g0[3] = (u32)((ga >> 32) & 0x01FFFFFFu) | (2u << 30); // global_addr[56:32], type=2
  const u32 td0 = 32, td1 = (u32)DD;                 // tensor dims (elements)
  const u32 tl0 = 32, tl1 = (u32)DD;                 // tile dims
  const u64 st0 = (u64)(u32)Kd;                      // dim0 stride (elements)
  i32x8 g1;
  g1[0] = (int)(1u << 16);                           // data_size = 1 -> 2 bytes
  g1[1] = (int)((td0 & 0xFFFFu) << 16);              // tensor_dim0[15:0]  @ bits 63:48
  g1[2] = (int)((td0 >> 16) | ((td1 & 0xFFFFu) << 16)); // dim0[31:16] | dim1[15:0]
  g1[3] = (int)((td1 >> 16) | (tl0 << 16));          // dim1[31:16] | tile_dim0
  g1[4] = (int)(tl1);                                // tile_dim1 (tile_dim2 = 0)
  g1[5] = (int)(u32)(st0 & 0xFFFFFFFFu);             // tensor_dim0_stride[31:0]
  g1[6] = (int)(u32)((st0 >> 32) & 0xFFFFu);         // stride[47:32] (dim1 stride = 0)
  g1[7] = 0;
  i32x4 z4 = (i32x4){0, 0, 0, 0};
  i32x8 z8 = (i32x8){0, 0, 0, 0, 0, 0, 0, 0};
  __builtin_amdgcn_tensor_load_to_lds(g0, g1, z4, z4, z8, 0);
}
#endif

// ---------------------------------------------------------------------------
// Projection GEMM (Ncols fixed to DD=192):  C[m][n] = sum_k A[m][k]*Bt[n][k] + bias[n]
// Block = 4 waves = 64 rows; each wave owns a 16x192 strip (12 accumulators),
// so A is streamed exactly once.  B k-slice double-buffered in LDS via TDM.
// ---------------------------------------------------------------------------
__global__ __launch_bounds__(128) void k_wmma_gemm_kv(
    const u16* __restrict__ A, const u16* __restrict__ Bt,
    const float* __restrict__ bias, float* __restrict__ outF, int Kd) {
  __shared__ __align__(16) u16 bslice[2][DD * 32];   // 2 x 12 KB double buffer

  const int tid  = threadIdx.x;
  const int wave = tid >> 5;
  const int lane = tid & 31;
  const int half = lane >> 4, lid = lane & 15;
  const int rowBase = blockIdx.x * 64 + wave * 16;
  const u16* arow = A + (size_t)(rowBase + lid) * Kd;

  v8f acc[12];
#pragma unroll
  for (int t = 0; t < 12; ++t) acc[t] = (v8f){};

  const int steps = Kd / 32;

#if HAVE_TDM
  if (wave == 0) tdm_stage_b(Bt, Kd, 0, (u32)(uintptr_t)&bslice[0][0]);
#endif

  for (int s = 0; s < steps; ++s) {
    const int k0 = s * 32;
#if HAVE_TDM
    if (wave == 0) {
      if (s + 1 < steps) {
        tdm_stage_b(Bt, Kd, k0 + 32, (u32)(uintptr_t)&bslice[(s + 1) & 1][0]);
        // only the just-issued (next) TDM may remain -> current slice is ready
        __builtin_amdgcn_s_wait_tensorcnt(1);
      } else {
        __builtin_amdgcn_s_wait_tensorcnt(0);
      }
    }
    __syncthreads();
#else
    // fallback: cooperative global->LDS copy of the B slice
    __syncthreads();
    {
      const u32* src32 = (const u32*)Bt;
      u32* dst32 = (u32*)&bslice[s & 1][0];
      for (int i = tid; i < DD * 16; i += 128) {
        int n = i >> 4, c = i & 15;
        dst32[i] = src32[(size_t)n * (Kd / 2) + (k0 / 2) + c];
      }
    }
    __syncthreads();
#endif

    Frag a;
    a.q[0] = *(const V4*)(arow + k0 + 16 * half);
    a.q[1] = *(const V4*)(arow + k0 + 16 * half + 8);
    const u16* bs = &bslice[s & 1][0];
#pragma unroll
    for (int tn = 0; tn < 12; ++tn) {
      Frag b;
      const u16* brow = bs + (tn * 16 + lid) * 32 + 16 * half;
      b.q[0] = *(const V4*)(brow);
      b.q[1] = *(const V4*)(brow + 8);
      acc[tn] = __builtin_amdgcn_wmma_f32_16x16x32_bf16(false, a.bf, false, b.bf,
                                                        (short)0, acc[tn], false, false);
    }
    __syncthreads();   // all waves done with bslice[s&1] before TDM rewrites it
  }

#pragma unroll
  for (int tn = 0; tn < 12; ++tn) {
    int n = tn * 16 + lid;
    float bv = bias ? bias[n] : 0.f;
#pragma unroll
    for (int v = 0; v < 8; ++v) {
      int m = rowBase + v + 8 * half;
      outF[(size_t)m * DD + n] = acc[tn][v] + bv;
    }
  }
}

// ---------------------------------------------------------------------------
// Generic WMMA GEMM (one wave per 16x16 tile) for the small epilogue matmuls.
// Output either f32 (outF) or bf16 (outBF) at [m*ldo + colOff + n].
// ---------------------------------------------------------------------------
__global__ __launch_bounds__(32) void k_wmma_gemm(
    const u16* __restrict__ A, const u16* __restrict__ Bt,
    const float* __restrict__ bias,
    float* __restrict__ outF, u16* __restrict__ outBF,
    int Kd, int ldo, int colOff, int relu) {
  const int lane = threadIdx.x;
  const int half = lane >> 4;
  const int lid  = lane & 15;
  const int tm = blockIdx.y, tn = blockIdx.x;

  const u16* arow = A  + (size_t)(tm * 16 + lid) * Kd;
  const u16* brow = Bt + (size_t)(tn * 16 + lid) * Kd;

  v8f acc = {};
  for (int k0 = 0; k0 < Kd; k0 += 32) {
    Frag a, b;
    a.q[0] = *(const V4*)(arow + k0 + 16 * half);
    a.q[1] = *(const V4*)(arow + k0 + 16 * half + 8);
    b.q[0] = *(const V4*)(brow + k0 + 16 * half);
    b.q[1] = *(const V4*)(brow + k0 + 16 * half + 8);
    acc = __builtin_amdgcn_wmma_f32_16x16x32_bf16(false, a.bf, false, b.bf,
                                                  (short)0, acc, false, false);
  }

  const int n = tn * 16 + lid;
  const float bv = bias ? bias[n] : 0.f;
#pragma unroll
  for (int v = 0; v < 8; ++v) {
    int m = tm * 16 + v + 8 * half;
    float val = acc[v] + bv;
    if (relu) val = fmaxf(val, 0.f);
    size_t off = (size_t)m * ldo + colOff + n;
    if (outF) outF[off] = val;
    else      outBF[off] = f2bf(val);
  }
}

// ---------------------------------------------------------------------------
// Repack projections to per-head bf16 layouts.
//   qbf/kbf : [H][N][96]   vtbf : [H][96][N] (V transposed: PV B-operand
//   becomes contraction-major so each lane reads one contiguous 32B frag)
// ---------------------------------------------------------------------------
__global__ void k_repack_qk(const float* __restrict__ qF, const float* __restrict__ kF,
                            u16* __restrict__ qbf, u16* __restrict__ kbf) {
  long idx = (long)blockIdx.x * blockDim.x + threadIdx.x;
  long total = (long)NN * DD;
  if (idx >= total) return;
  int n = (int)(idx / DD), j = (int)(idx % DD);
  int h = j / DHEAD, d = j % DHEAD;
  long dst = ((long)h * NN + n) * DHEAD + d;
  qbf[dst] = f2bf(qF[idx]);
  kbf[dst] = f2bf(kF[idx]);
}

__global__ void k_repack_vt(const float* __restrict__ vF, u16* __restrict__ vtbf) {
  long idx = (long)blockIdx.x * blockDim.x + threadIdx.x;
  long total = (long)NN * DD;
  if (idx >= total) return;
  int n = (int)(idx / DD), j = (int)(idx % DD);
  int h = j / DHEAD, d = j % DHEAD;
  vtbf[((long)h * DHEAD + d) * NN + n] = f2bf(vF[idx]);
}

// ---------------------------------------------------------------------------
// Flash attention: one wave owns (head h, 16-query tile).  Streams keys in
// chunks of 32: 6 WMMAs for S = Q K^T, shuffle-reduced online softmax,
// P staged through LDS (C-layout -> A-layout transpose), 6 WMMAs for O += P V.
// ---------------------------------------------------------------------------
__global__ __launch_bounds__(32) void k_flash_attn(
    const u16* __restrict__ qbf,   // [H][N][96]
    const u16* __restrict__ kbf,   // [H][N][96]
    const u16* __restrict__ vtbf,  // [H][96][N]
    u16* __restrict__ obf) {       // [N][192]  (heads concatenated)
  __shared__ __align__(16) u16 plds[16 * 32];   // 1KB P staging tile

  const int lane = threadIdx.x;
  const int half = lane >> 4;
  const int lid  = lane & 15;
  const int nQt  = NN / 16;
  const int h  = blockIdx.x / nQt;
  const int qt = blockIdx.x % nQt;
  const float scale = 0.10206207262f;           // 1/sqrt(96)

  Frag qa[3];
  const u16* qrow = qbf + ((size_t)h * NN + qt * 16 + lid) * DHEAD;
#pragma unroll
  for (int t = 0; t < 3; ++t) {
    qa[t].q[0] = *(const V4*)(qrow + 32 * t + 16 * half);
    qa[t].q[1] = *(const V4*)(qrow + 32 * t + 16 * half + 8);
  }

  v8f oacc[6];
#pragma unroll
  for (int t = 0; t < 6; ++t) oacc[t] = (v8f){};
  float rm[8], rl[8];
#pragma unroll
  for (int v = 0; v < 8; ++v) { rm[v] = -1e30f; rl[v] = 0.f; }

  for (int kc = 0; kc < NN; kc += 32) {
    if (kc + 32 < NN) {  // prefetch next key chunk (lowers to global_prefetch_b8)
      __builtin_prefetch(kbf + ((size_t)h * NN + kc + 32 + lid) * DHEAD, 0, 1);
      __builtin_prefetch(vtbf + ((size_t)h * DHEAD + lid) * NN + kc + 32, 0, 1);
    }

    // --- S = Q K^T for two 16-key tiles ---
    float s0[8], s1[8];
#pragma unroll
    for (int sub = 0; sub < 2; ++sub) {
      const u16* krow = kbf + ((size_t)h * NN + kc + sub * 16 + lid) * DHEAD;
      v8f acc = {};
#pragma unroll
      for (int t = 0; t < 3; ++t) {
        Frag kb;
        kb.q[0] = *(const V4*)(krow + 32 * t + 16 * half);
        kb.q[1] = *(const V4*)(krow + 32 * t + 16 * half + 8);
        acc = __builtin_amdgcn_wmma_f32_16x16x32_bf16(false, qa[t].bf, false, kb.bf,
                                                      (short)0, acc, false, false);
      }
      float* s = sub ? s1 : s0;
#pragma unroll
      for (int v = 0; v < 8; ++v) s[v] = acc[v] * scale;
    }

    // --- online softmax (row m = v + 8*half lives across one 16-lane group) ---
    float alpha[8];
#pragma unroll
    for (int v = 0; v < 8; ++v) {
      float mx = fmaxf(s0[v], s1[v]);
      mx = fmaxf(mx, __shfl_xor(mx, 1, 32));
      mx = fmaxf(mx, __shfl_xor(mx, 2, 32));
      mx = fmaxf(mx, __shfl_xor(mx, 4, 32));
      mx = fmaxf(mx, __shfl_xor(mx, 8, 32));
      float mnew = fmaxf(rm[v], mx);
      float a  = __expf(rm[v] - mnew);
      float p0 = __expf(s0[v] - mnew);
      float p1 = __expf(s1[v] - mnew);
      float ls = p0 + p1;
      ls += __shfl_xor(ls, 1, 32);
      ls += __shfl_xor(ls, 2, 32);
      ls += __shfl_xor(ls, 4, 32);
      ls += __shfl_xor(ls, 8, 32);
      rl[v] = rl[v] * a + ls;
      rm[v] = mnew;
      alpha[v] = a;
      int m = v + 8 * half;
      plds[m * 32 + lid]      = f2bf(p0);
      plds[m * 32 + 16 + lid] = f2bf(p1);
    }
#pragma unroll
    for (int t = 0; t < 6; ++t)
#pragma unroll
      for (int v = 0; v < 8; ++v) oacc[t][v] *= alpha[v];

    __syncthreads();   // single-wave WG: C->A transpose fence through LDS

    Frag pa;
    const u16* prow = plds + lid * 32;
    pa.q[0] = *(const V4*)(prow + 16 * half);
    pa.q[1] = *(const V4*)(prow + 16 * half + 8);
#pragma unroll
    for (int t = 0; t < 6; ++t) {
      const u16* vrow = vtbf + ((size_t)h * DHEAD + t * 16 + lid) * NN + kc + 16 * half;
      Frag vb;
      vb.q[0] = *(const V4*)(vrow);
      vb.q[1] = *(const V4*)(vrow + 8);
      oacc[t] = __builtin_amdgcn_wmma_f32_16x16x32_bf16(false, pa.bf, false, vb.bf,
                                                        (short)0, oacc[t], false, false);
    }
    __syncthreads();
  }

  float inv[8];
#pragma unroll
  for (int v = 0; v < 8; ++v) inv[v] = 1.0f / rl[v];
#pragma unroll
  for (int t = 0; t < 6; ++t) {
#pragma unroll
    for (int v = 0; v < 8; ++v) {
      int m   = qt * 16 + v + 8 * half;
      int col = h * DHEAD + t * 16 + lid;
      obf[(size_t)m * DD + col] = f2bf(oacc[t][v] * inv[v]);
    }
  }
}

// ---------------------------------------------------------------------------
// Host launch
// ---------------------------------------------------------------------------
extern "C" void kernel_launch(void* const* d_in, const int* in_sizes, int n_in,
                              void* d_out, int out_size, void* d_ws, size_t ws_size,
                              hipStream_t stream) {
  const float* features = (const float*)d_in[0];
  const float* edgef    = (const float*)d_in[1];
  const float* timef    = (const float*)d_in[2];
  const int*   neigh    = (const int*)d_in[3];
  const int*   nidx     = (const int*)d_in[4];
  const float* Wq = (const float*)d_in[5];  const float* bq = (const float*)d_in[6];
  const float* Wk = (const float*)d_in[7];  const float* bk = (const float*)d_in[8];
  const float* Wv = (const float*)d_in[9];  const float* bv = (const float*)d_in[10];
  const float* Wo = (const float*)d_in[11]; const float* bo = (const float*)d_in[12];
  const float* W1 = (const float*)d_in[13]; const float* b1 = (const float*)d_in[14];
  const float* W2 = (const float*)d_in[15]; const float* b2 = (const float*)d_in[16];
  float* out = (float*)d_out;
  (void)in_sizes; (void)n_in; (void)out_size;

  char* p = (char*)d_ws;
  auto alloc = [&](size_t bytes) -> char* {
    char* r = p; p += (bytes + 255) & ~(size_t)255; return r;
  };
  u16* agg  = (u16*)alloc((size_t)NN * KDIM * 2);
  u16* qin  = (u16*)alloc((size_t)NN * DD * 2);
  u16* hbuf = (u16*)alloc((size_t)NN * DHE * 2);
  u16* WqT  = (u16*)alloc((size_t)DD * DD * 2);
  u16* WkT  = (u16*)alloc((size_t)DD * KDIM * 2);
  u16* WvT  = (u16*)alloc((size_t)DD * KDIM * 2);
  u16* WoT  = (u16*)alloc((size_t)DD * DD * 2);
  u16* W1T  = (u16*)alloc((size_t)EE * DHE * 2);
  u16* W2T  = (u16*)alloc((size_t)EE * EE * 2);
  float* qF = (float*)alloc((size_t)NN * DD * 4);
  float* kF = (float*)alloc((size_t)NN * DD * 4);
  float* vF = (float*)alloc((size_t)NN * DD * 4);
  u16* qbf  = (u16*)alloc((size_t)HH * NN * DHEAD * 2);
  u16* kbf  = (u16*)alloc((size_t)HH * NN * DHEAD * 2);
  u16* vtbf = (u16*)alloc((size_t)HH * DHEAD * NN * 2);
  u16* obf  = (u16*)alloc((size_t)NN * DD * 2);
  u16* rbf  = (u16*)alloc((size_t)NN * EE * 2);
  if ((size_t)(p - (char*)d_ws) > ws_size) return;  // insufficient workspace

  const int TB = 256;
  auto gs = [](long total, int tb) { return (unsigned)((total + tb - 1) / tb); };

  // prep
  k_build_agg<<<gs((long)NN * KDIM, TB), TB, 0, stream>>>(features, edgef, timef, neigh, agg);
  k_build_qin_h<<<gs((long)NN * DD, TB), TB, 0, stream>>>(features, nidx, qin, hbuf);
  k_transpose_w<<<gs((long)DD * DD, TB),   TB, 0, stream>>>(Wq, WqT, DD, DD);
  k_transpose_w<<<gs((long)KDIM * DD, TB), TB, 0, stream>>>(Wk, WkT, KDIM, DD);
  k_transpose_w<<<gs((long)KDIM * DD, TB), TB, 0, stream>>>(Wv, WvT, KDIM, DD);
  k_transpose_w<<<gs((long)DD * DD, TB),   TB, 0, stream>>>(Wo, WoT, DD, DD);
  k_transpose_w<<<gs((long)DHE * EE, TB),  TB, 0, stream>>>(W1, W1T, DHE, EE);
  k_transpose_w<<<gs((long)EE * EE, TB),   TB, 0, stream>>>(W2, W2T, EE, EE);

  // projections: row-strip WMMA GEMM with TDM-staged B slices (bias folded)
  k_wmma_gemm_kv<<<NN / 64, 128, 0, stream>>>(qin, WqT, bq, qF, DD);
  k_wmma_gemm_kv<<<NN / 64, 128, 0, stream>>>(agg, WkT, bk, kF, KDIM);
  k_wmma_gemm_kv<<<NN / 64, 128, 0, stream>>>(agg, WvT, bv, vF, KDIM);

  // repack to per-head layouts
  k_repack_qk<<<gs((long)NN * DD, TB), TB, 0, stream>>>(qF, kF, qbf, kbf);
  k_repack_vt<<<gs((long)NN * DD, TB), TB, 0, stream>>>(vF, vtbf);

  // flash attention
  k_flash_attn<<<HH * (NN / 16), 32, 0, stream>>>(qbf, kbf, vtbf, obf);

  // Wo: attn_out (bf16) written into hbuf columns [128:320) -> h = concat(x, attn_out)
  dim3 gQ(DD / 16, NN / 16);
  k_wmma_gemm<<<gQ, 32, 0, stream>>>(obf, WoT, bo, nullptr, hbuf, DD, DHE, EE, 0);

  // MLP layer 1 (ReLU, bf16 out) and layer 2 (f32 -> d_out)
  dim3 gM(EE / 16, NN / 16);
  k_wmma_gemm<<<gM, 32, 0, stream>>>(hbuf, W1T, b1, nullptr, rbf, DHE, EE, 0, 1);
  k_wmma_gemm<<<gM, 32, 0, stream>>>(rbf, W2T, b2, out, nullptr, EE, EE, 0, 0);
}
